// VOXCPM_FEAT_DECODER_37151467110419
// MI455X (gfx1250) — matrix operand order repaired and checked by reading, not compiled, not run on hardware
//
#include <hip/hip_runtime.h>
#include <hip/hip_bf16.h>
#include <math.h>

#define HIDDEN 1024
#define HEADS 16
#define HEAD_DIM 64
#define FFN 4096
#define LAYERS 8
#define QLEN 9
#define ROWS 18
#define LATENT 64

typedef __attribute__((ext_vector_type(16))) __bf16 v16bf;
typedef __attribute__((ext_vector_type(8)))  float  v8f;
typedef __attribute__((ext_vector_type(4)))  float  v4f;

// ---------------- WMMA fragment loaders (fp32 memory -> bf16 fragments) ----------------

__device__ __forceinline__ v16bf frag_zero() {
    v16bf r;
#pragma unroll
    for (int i = 0; i < 16; ++i) r[i] = (__bf16)0.0f;
    return r;
}

// A fragment 16x32 (MxK) bf16.  lane<16: M=lane, K slots {kb+0..7, kb+16..23}
//                               lane>=16: M=lane-16, K slots {kb+8..15, kb+24..31}
__device__ __forceinline__ v16bf load_a(const float* __restrict__ src, int ld, int kb,
                                        int row0, int mlimit) {
    const int lane = threadIdx.x & 31;
    const int m = row0 + (lane & 15);
    v16bf r = frag_zero();
    if (m < mlimit) {
        const v4f* p = (const v4f*)(src + (size_t)m * ld + kb + ((lane & 16) ? 8 : 0));
        v4f x0 = p[0], x1 = p[1];      // K = kb(+8) + 0..7
        v4f x2 = p[4], x3 = p[5];      // K = kb(+8) + 16..23
#pragma unroll
        for (int i = 0; i < 4; ++i) {
            r[i]      = (__bf16)x0[i];
            r[4 + i]  = (__bf16)x1[i];
            r[8 + i]  = (__bf16)x2[i];
            r[12 + i] = (__bf16)x3[i];
        }
    }
    return r;
}

// B fragment 32x16 (KxN) bf16 from W[N,K] row-major (so D = A @ W^T).
// lane<16: N=lane, K={kb+0..7, kb+16..23} ;  lane>=16: N=lane-16, K={kb+8..15, kb+24..31}
__device__ __forceinline__ v16bf load_b(const float* __restrict__ w, int ld, int kb, int ntile) {
    const int lane = threadIdx.x & 31;
    const v4f* p = (const v4f*)(w + (size_t)(ntile * 16 + (lane & 15)) * ld + kb +
                                ((lane & 16) ? 8 : 0));
    v4f x0 = p[0], x1 = p[1], x2 = p[4], x3 = p[5];
    v16bf r;
#pragma unroll
    for (int i = 0; i < 4; ++i) {
        r[i]      = (__bf16)x0[i];
        r[4 + i]  = (__bf16)x1[i];
        r[8 + i]  = (__bf16)x2[i];
        r[12 + i] = (__bf16)x3[i];
    }
    return r;
}

// ---------------- GEMM: out[18,N] = act[18,K] @ W[N,K]^T (+bias) (+res) ----------------
// Block = 128 threads = 4 waves. One block per 16-wide N tile; 4-way split-K, LDS reduce.
__global__ void gemm_wmma(const float* __restrict__ act, const float* __restrict__ w,
                          const float* bias, const float* res, float* out, int K, int N) {
    __shared__ float red[4 * 512];
    const int wave   = threadIdx.x >> 5;
    const int lane   = threadIdx.x & 31;
    const int ntile  = blockIdx.x;
    const int kchunk = K >> 2;               // >=256, multiple of 32
    const int k0     = wave * kchunk;

    v8f acc0 = {};
    v8f acc1 = {};
    for (int k = k0; k < k0 + kchunk; k += 32) {
        v16bf b  = load_b(w, K, k, ntile);
        v16bf a0 = load_a(act, K, k, 0, ROWS);
        v16bf a1 = load_a(act, K, k, 16, ROWS);
        acc0 = __builtin_amdgcn_wmma_f32_16x16x32_bf16(false, a0, false, b, (short)0, acc0,
                                                       false, false);
        acc1 = __builtin_amdgcn_wmma_f32_16x16x32_bf16(false, a1, false, b, (short)0, acc1,
                                                       false, false);
    }
    float* dst = &red[wave * 512];
#pragma unroll
    for (int j = 0; j < 8; ++j) {
        dst[j * 32 + lane]       = acc0[j];
        dst[256 + j * 32 + lane] = acc1[j];
    }
    __syncthreads();
    for (int idx = threadIdx.x; idx < 512; idx += 128) {
        float s = red[idx] + red[512 + idx] + red[1024 + idx] + red[1536 + idx];
        int tile = idx >> 8, j = (idx >> 5) & 7, l = idx & 31;
        int m = tile * 16 + j + ((l & 16) ? 8 : 0);
        int n = ntile * 16 + (l & 15);
        if (m < ROWS) {
            if (bias) s += bias[n];
            if (res)  s += res[(size_t)m * N + n];
            out[(size_t)m * N + n] = s;
        }
    }
}

// ---------------- prep: build hs[18,1024] ----------------
__global__ void prep_kernel(const int* step, const float* __restrict__ random,
                            const float* __restrict__ dit, const float* __restrict__ feat,
                            const float* __restrict__ t_emb, const float* __restrict__ in_w,
                            const float* __restrict__ in_b, float* __restrict__ hs) {
    const int m = blockIdx.x, b = m / QLEN, t = m % QLEN;
    const float* te = t_emb + (size_t)(*step) * HIDDEN;
    for (int c = threadIdx.x; c < HIDDEN; c += blockDim.x) {
        float val;
        if (t == 0) {
            val = te[c] + (b == 0 ? dit[c] : 0.0f);
        } else if (t <= 4) {
            val = feat[(size_t)(b * 4 + (t - 1)) * HIDDEN + c];
        } else {
            const int r = t - 5;
            float s = in_b[c];
            const float* wr = in_w + (size_t)c * LATENT;
            const float* rr = random + (size_t)r * LATENT;
#pragma unroll 4
            for (int e = 0; e < LATENT; ++e) s += rr[e] * wr[e];
            val = s;
        }
        hs[(size_t)m * HIDDEN + c] = val;
    }
}

// ---------------- rmsnorm: y = x * rsqrt(sum(x^2)) per row ----------------
__global__ void rmsnorm_kernel(const float* __restrict__ x, float* __restrict__ y) {
    __shared__ float sm[256];
    const int m = blockIdx.x;
    float p = 0.0f;
    for (int c = threadIdx.x; c < HIDDEN; c += 256) {
        float v = x[(size_t)m * HIDDEN + c];
        p += v * v;
    }
    sm[threadIdx.x] = p;
    __syncthreads();
    for (int s = 128; s > 0; s >>= 1) {
        if (threadIdx.x < s) sm[threadIdx.x] += sm[threadIdx.x + s];
        __syncthreads();
    }
    const float sc = rsqrtf(sm[0]);
    for (int c = threadIdx.x; c < HIDDEN; c += 256)
        y[(size_t)m * HIDDEN + c] = x[(size_t)m * HIDDEN + c] * sc;
}

// ---------------- attention: RoPE + softmax(QK^T)V, per (batch, head) ----------------
__global__ void attn_kernel(const float* __restrict__ qkv, float* __restrict__ o) {
    __shared__ float q[QLEN][64], k[QLEN][64], v[QLEN][64];
    __shared__ float sc[QLEN][QLEN + 1], at[QLEN][QLEN + 1];
    const int b = blockIdx.x >> 4, h = blockIdx.x & 15;
    const int d = threadIdx.x;  // 0..63
    for (int t = 0; t < QLEN; ++t) {
        size_t base = (size_t)((b * QLEN + t) * 48) * 64;
        q[t][d] = qkv[base + (size_t)h * 64 + d];
        k[t][d] = qkv[base + (size_t)(16 + h) * 64 + d];
        v[t][d] = qkv[base + (size_t)(32 + h) * 64 + d];
    }
    __syncthreads();
    // RoPE: inv = 10000^(-(d%32)/32); rot = concat([x2, x1]); sin sign flips on first half
    const int j = d & 31;
    const float inv = expf(-(float)j * 0.28782313662425572f);  // ln(10000)/32
    const int pd = (d < 32) ? d + 32 : d - 32;
    float qr[QLEN], kr[QLEN];
    for (int t = 0; t < QLEN; ++t) {
        float ang = (float)t * inv;
        float c = cosf(ang), s = sinf(ang);
        float ss = (d < 32) ? -s : s;
        qr[t] = q[t][d] * c + q[t][pd] * ss;
        kr[t] = k[t][d] * c + k[t][pd] * ss;
    }
    __syncthreads();
    for (int t = 0; t < QLEN; ++t) { q[t][d] = qr[t]; k[t][d] = kr[t]; }
    __syncthreads();
    for (int p = d; p < QLEN * QLEN; p += 64) {
        int qi = p / QLEN, ki = p % QLEN;
        float s = 0.0f;
#pragma unroll 8
        for (int e = 0; e < 64; ++e) s += q[qi][e] * k[ki][e];
        sc[qi][ki] = s;   // no 1/sqrt(d) scale in reference
    }
    __syncthreads();
    if (d < QLEN) {
        float mx = -1e30f;
        for (int ki = 0; ki < QLEN; ++ki) mx = fmaxf(mx, sc[d][ki]);
        float sum = 0.0f;
        for (int ki = 0; ki < QLEN; ++ki) {
            float e = expf(sc[d][ki] - mx);
            at[d][ki] = e;
            sum += e;
        }
        float r = 1.0f / sum;
        for (int ki = 0; ki < QLEN; ++ki) at[d][ki] *= r;
    }
    __syncthreads();
    for (int qi = 0; qi < QLEN; ++qi) {
        float s = 0.0f;
        for (int ki = 0; ki < QLEN; ++ki) s += at[qi][ki] * v[ki][d];
        o[(size_t)(b * QLEN + qi) * HIDDEN + (size_t)h * 64 + d] = s;
    }
}

// ---------------- SwiGLU: ffn = silu(g) * u ----------------
__global__ void swiglu_kernel(const float* __restrict__ gu, float* __restrict__ fi) {
    int idx = blockIdx.x * blockDim.x + threadIdx.x;
    if (idx < ROWS * FFN) {
        int m = idx / FFN, jj = idx - m * FFN;
        float g = gu[(size_t)m * (2 * FFN) + jj];
        float u = gu[(size_t)m * (2 * FFN) + FFN + jj];
        fi[idx] = (g / (1.0f + expf(-g))) * u;
    }
}

// ---------------- final: rms -> out_proj -> CFG -> random + guided*dt ----------------
__global__ void final_kernel(const float* __restrict__ hs, const float* __restrict__ wout,
                             const float* __restrict__ bout, const float* __restrict__ random,
                             const float* cfg, const float* cfgm, const float* dt,
                             const int* step, float* __restrict__ out) {
    __shared__ float sm[512];
    __shared__ float scale[8];
    __shared__ float obuf[512];
    __shared__ float dotsq[2];
    const int tid = threadIdx.x;
    // per-row sum of squares (8 rows: (b, 5+r)), 64 threads each
    {
        int rrow = tid >> 6, jj = tid & 63;
        int b = rrow >> 2, r = rrow & 3;
        const float* xr = hs + (size_t)(b * QLEN + 5 + r) * HIDDEN;
        float p = 0.0f;
        for (int c = jj; c < HIDDEN; c += 64) { float v = xr[c]; p += v * v; }
        sm[tid] = p;
    }
    __syncthreads();
    for (int s = 32; s > 0; s >>= 1) {
        if ((tid & 63) < s) sm[tid] += sm[tid + s];
        __syncthreads();
    }
    if ((tid & 63) == 0) scale[tid >> 6] = rsqrtf(sm[tid]);
    __syncthreads();
    // out projection: tid -> (b, r, c)
    {
        int b = tid >> 8, idx = tid & 255, r = idx >> 6, c = idx & 63;
        const float* xr = hs + (size_t)(b * QLEN + 5 + r) * HIDDEN;
        const float* wr = wout + (size_t)c * HIDDEN;
        float s = 0.0f;
#pragma unroll 4
        for (int e = 0; e < HIDDEN; ++e) s += xr[e] * wr[e];
        obuf[tid] = s * scale[b * 4 + r] + bout[c];
    }
    __syncthreads();
    // dot(pos,neg) and |neg|^2 over 256 elements
    if (tid < 256) sm[tid] = obuf[tid] * obuf[256 + tid];
    else           sm[tid] = obuf[tid] * obuf[tid];
    __syncthreads();
    for (int s = 128; s > 0; s >>= 1) {
        if (tid < s) sm[tid] += sm[tid + s];
        if (tid >= 256 && tid < 256 + s) sm[tid] += sm[tid + s];
        __syncthreads();
    }
    if (tid == 0) { dotsq[0] = sm[0]; dotsq[1] = sm[256]; }
    __syncthreads();
    if (tid < 256) {
        float coef = cfgm[0] * (dotsq[0] / dotsq[1]);
        float guided = cfg[0] * obuf[tid] + coef * obuf[256 + tid];
        out[tid] = random[tid] + guided * dt[*step];
    }
}

// ---------------- launch ----------------
extern "C" void kernel_launch(void* const* d_in, const int* in_sizes, int n_in,
                              void* d_out, int out_size, void* d_ws, size_t ws_size,
                              hipStream_t stream) {
    const int*   step   = (const int*)  d_in[0];
    const float* random = (const float*)d_in[1];
    const float* dit    = (const float*)d_in[2];
    const float* feat   = (const float*)d_in[3];
    const float* cfg    = (const float*)d_in[4];
    const float* cfgm   = (const float*)d_in[5];
    const float* t_emb  = (const float*)d_in[6];
    const float* dt     = (const float*)d_in[7];
    const float* in_w   = (const float*)d_in[8];
    const float* in_b   = (const float*)d_in[9];
    const float* qkv_w  = (const float*)d_in[10];
    const float* qkv_b  = (const float*)d_in[11];
    const float* o_w    = (const float*)d_in[12];
    const float* gu_w   = (const float*)d_in[13];
    const float* dn_w   = (const float*)d_in[14];
    const float* out_w  = (const float*)d_in[15];
    const float* out_b  = (const float*)d_in[16];

    float* ws   = (float*)d_ws;
    float* hs   = ws;                              // 18*1024
    float* nrm  = hs   + ROWS * HIDDEN;            // 18*1024
    float* qkv  = nrm  + ROWS * HIDDEN;            // 18*3072
    float* aout = qkv  + ROWS * 3 * HIDDEN;        // 18*1024
    float* gu   = aout + ROWS * HIDDEN;            // 18*8192
    float* ffn  = gu   + ROWS * 2 * FFN;           // 18*4096

    prep_kernel<<<ROWS, 256, 0, stream>>>(step, random, dit, feat, t_emb, in_w, in_b, hs);
    for (int i = 0; i < LAYERS; ++i) {
        rmsnorm_kernel<<<ROWS, 256, 0, stream>>>(hs, nrm);
        gemm_wmma<<<(3 * HIDDEN) / 16, 128, 0, stream>>>(
            nrm, qkv_w + (size_t)i * 3 * HIDDEN * HIDDEN, qkv_b + (size_t)i * 3 * HIDDEN,
            nullptr, qkv, HIDDEN, 3 * HIDDEN);
        attn_kernel<<<32, 64, 0, stream>>>(qkv, aout);
        gemm_wmma<<<HIDDEN / 16, 128, 0, stream>>>(
            aout, o_w + (size_t)i * HIDDEN * HIDDEN, nullptr, hs, hs, HIDDEN, HIDDEN);
        rmsnorm_kernel<<<ROWS, 256, 0, stream>>>(hs, nrm);
        gemm_wmma<<<(2 * FFN) / 16, 128, 0, stream>>>(
            nrm, gu_w + (size_t)i * 2 * FFN * HIDDEN, nullptr, nullptr, gu, HIDDEN, 2 * FFN);
        swiglu_kernel<<<(ROWS * FFN + 255) / 256, 256, 0, stream>>>(gu, ffn);
        gemm_wmma<<<HIDDEN / 16, 128, 0, stream>>>(
            ffn, dn_w + (size_t)i * HIDDEN * FFN, nullptr, hs, hs, FFN, HIDDEN);
    }
    final_kernel<<<1, 512, 0, stream>>>(hs, out_w, out_b, random, cfg, cfgm, dt, step,
                                        (float*)d_out);
}
